// QuantizedCNN_80564996539186
// MI455X (gfx1250) — 8-bit matrix operand order repaired and checked by reading, compiled, not run on hardware
//
#include <hip/hip_runtime.h>
#include <hip/hip_bf16.h>


typedef int v8i __attribute__((ext_vector_type(8)));

__device__ __forceinline__ int trunc24(int v) {
    v = v < 0 ? 0 : v;
    v = v > 0x7FFFFF ? 0x7FFFFF : v;
    return v & 255;
}

__device__ __forceinline__ int a_kbase(int v, int h) {
    return ((v >> 1) << 4) + (h << 3) + ((v & 1) << 2);
}
__device__ __forceinline__ int b_kbase(int v, int h) {
    return ((v >> 1) << 4) + (h << 3) + ((v & 1) << 2);
}

__global__ __launch_bounds__(32)
void qcnn_fused_wmma(const int* __restrict__ x,
                     const int* __restrict__ w1, const int* __restrict__ b1,
                     const int* __restrict__ w2, const int* __restrict__ b2,
                     const int* __restrict__ wf1, const int* __restrict__ bf1,
                     const int* __restrict__ wf2, const int* __restrict__ bf2,
                     float* __restrict__ out)
{
    __shared__ __align__(16) unsigned char xs[16 * 256];
    __shared__ __align__(16) unsigned char a1[16 * 2080];
    __shared__ __align__(16) unsigned char a2[16 * 192];
    __shared__ __align__(16) unsigned char a3[16 * 16];
    __shared__ int B1L[10][32][8];
    __shared__ int B2L[8][32][8];
    __shared__ int BF1L[3][32][8];

    const int lane = threadIdx.x;
    const int m    = lane & 15;
    const int h    = lane >> 4;
    const int s0   = blockIdx.x * 16;

    {
        const int gbase = s0 * 240;
        for (int t = lane; t < 16 * 240; t += 32) {
            int val = x[gbase + t];
            xs[(t / 240) * 256 + (t % 240)] = (unsigned char)val;
        }
    }
    for (int t = lane; t < 16 * 15; t += 32) {
        int mm = t / 15, dwi = t % 15;
        *(int*)(a2 + mm * 192 + 132 + dwi * 4) = 0;
    }

    for (int c = 0; c < 10; ++c) {
        #pragma unroll
        for (int v = 0; v < 8; ++v) {
            unsigned int dw = 0;
            #pragma unroll
            for (int b = 0; b < 4; ++b) {
                int k = b_kbase(v, h) + b;
                int r = k >> 4, cc = k & 15;
                int dj = cc - m;
                unsigned int byte = 0;
                if (r < 3 && m < 14 && dj >= 0 && dj <= 2)
                    byte = (unsigned int)(unsigned char)(w1[c * 9 + r * 3 + dj]);
                dw |= byte << (8 * b);
            }
            B1L[c][lane][v] = (int)dw;
        }
    }
    for (int s = 0; s < 8; ++s) {
        #pragma unroll
        for (int v = 0; v < 8; ++v) {
            unsigned int dw = 0;
            #pragma unroll
            for (int b = 0; b < 4; ++b) {
                int k = s * 64 + b_kbase(v, h) + b;
                int rr = k >> 4, cc = k & 15;
                unsigned int byte = 0;
                if (rr < 30) {
                    int c = rr / 3, r = rr - c * 3;
                    int dj = cc - m;
                    if (m < 12 && dj >= 0 && dj <= 2)
                        byte = (unsigned int)(unsigned char)(w2[c * 9 + r * 3 + dj]);
                }
                dw |= byte << (8 * b);
            }
            B2L[s][lane][v] = (int)dw;
        }
    }
    for (int s = 0; s < 3; ++s) {
        #pragma unroll
        for (int v = 0; v < 8; ++v) {
            unsigned int dw = 0;
            #pragma unroll
            for (int b = 0; b < 4; ++b) {
                int k = s * 64 + b_kbase(v, h) + b;
                unsigned int byte = 0;
                if (k < 132 && m < 10)
                    byte = (unsigned int)(unsigned char)(wf1[m * 132 + k]);
                dw |= byte << (8 * b);
            }
            BF1L[s][lane][v] = (int)dw;
        }
    }
    __syncthreads();

    for (int i = 0; i < 13; ++i) {
        v8i A;
        #pragma unroll
        for (int v = 0; v < 8; ++v) {
            int ka = a_kbase(v, h);
            A[v] = (ka < 48) ? *(const int*)(xs + m * 256 + i * 16 + ka) : 0;
        }
        #pragma unroll
        for (int c = 0; c < 10; ++c) {
            v8i B;
            #pragma unroll
            for (int v = 0; v < 8; ++v) B[v] = B1L[c][lane][v];
            v8i acc = {0, 0, 0, 0, 0, 0, 0, 0};
            acc = __builtin_amdgcn_wmma_i32_16x16x64_iu8(false, A, true, B, acc, false, false);
            const int bias = b1[c];
            #pragma unroll
            for (int r = 0; r < 8; ++r) {
                int mm = r + 8 * h;
                unsigned char val = (m < 14) ? (unsigned char)trunc24(acc[r] + bias)
                                             : (unsigned char)0;
                a1[mm * 2080 + c * 208 + i * 16 + m] = val;
            }
        }
    }
    __syncthreads();

    for (int qi = 0; qi < 11; ++qi) {
        v8i acc = {0, 0, 0, 0, 0, 0, 0, 0};
        #pragma unroll
        for (int s = 0; s < 8; ++s) {
            v8i A;
            #pragma unroll
            for (int v = 0; v < 8; ++v) {
                int k = s * 64 + a_kbase(v, h);
                int dw = 0;
                if (k < 480) {
                    int rr = k >> 4, cc = k & 15;
                    int c = rr / 3, r = rr - c * 3;
                    dw = *(const int*)(a1 + m * 2080 + c * 208 + (qi + r) * 16 + cc);
                }
                A[v] = dw;
            }
            v8i B;
            #pragma unroll
            for (int v = 0; v < 8; ++v) B[v] = B2L[s][lane][v];
            acc = __builtin_amdgcn_wmma_i32_16x16x64_iu8(false, A, true, B, acc, false, false);
        }
        const int bias = b2[0];
        if (m < 12) {
            #pragma unroll
            for (int r = 0; r < 8; ++r) {
                int mm = r + 8 * h;
                a2[mm * 192 + qi * 12 + m] = (unsigned char)trunc24(acc[r] + bias);
            }
        }
    }
    __syncthreads();

    {
        v8i acc = {0, 0, 0, 0, 0, 0, 0, 0};
        #pragma unroll
        for (int s = 0; s < 3; ++s) {
            v8i A;
            #pragma unroll
            for (int v = 0; v < 8; ++v) {
                int k = s * 64 + a_kbase(v, h);
                A[v] = *(const int*)(a2 + m * 192 + k);
            }
            v8i B;
            #pragma unroll
            for (int v = 0; v < 8; ++v) B[v] = BF1L[s][lane][v];
            acc = __builtin_amdgcn_wmma_i32_16x16x64_iu8(false, A, true, B, acc, false, false);
        }
        if (m < 10) {
            const int bias = bf1[m];
            #pragma unroll
            for (int r = 0; r < 8; ++r)
                a3[(r + 8 * h) * 16 + m] = (unsigned char)trunc24(acc[r] + bias);
        }
    }
    __syncthreads();

    if (h == 0) {
        int acc2 = bf2[0];
        #pragma unroll
        for (int n = 0; n < 10; ++n)
            acc2 += (int)a3[m * 16 + n] * wf2[n];
        out[s0 + m] = (float)trunc24(acc2);
    }
}

extern "C" void kernel_launch(void* const* d_in, const int* in_sizes, int n_in,
                              void* d_out, int out_size, void* d_ws, size_t ws_size,
                              hipStream_t stream) {
    (void)n_in; (void)out_size; (void)d_ws; (void)ws_size;
    const int* x   = (const int*)d_in[0];
    const int* w1  = (const int*)d_in[1];
    const int* b1  = (const int*)d_in[2];
    const int* w2  = (const int*)d_in[3];
    const int* b2  = (const int*)d_in[4];
    const int* wf1 = (const int*)d_in[5];
    const int* bf1 = (const int*)d_in[6];
    const int* wf2 = (const int*)d_in[7];
    const int* bf2 = (const int*)d_in[8];
    float* out = (float*)d_out;

    const int nsamp = in_sizes[0] / 240;
    dim3 grid(nsamp / 16), block(32);
    qcnn_fused_wmma<<<grid, block, 0, stream>>>(x, w1, b1, w2, b2, wf1, bf1, wf2, bf2, out);
}
